// GraphModelWithMultipleVirtualNodes_72172630442242
// MI455X (gfx1250) — matrix operand order repaired and checked by reading, not compiled, not run on hardware
//
#include <hip/hip_runtime.h>
#include <hip/hip_bf16.h>

typedef __attribute__((ext_vector_type(16))) __bf16 v16bf;
typedef __attribute__((ext_vector_type(8)))  float  v8f;
typedef unsigned int u32x4 __attribute__((ext_vector_type(4)));
typedef int          i32x8 __attribute__((ext_vector_type(8)));
typedef int          i32x4 __attribute__((ext_vector_type(4)));

#define HD      1024
#define INDIM   128
#define OUTDIM  40
#define NVN     3
#define VH      128
#define GC      64
#define LN_EPS  1e-5f

#define TM 128
#define TN 128
#define TK 32

// ---------------------------------------------------------------------------
// LDS byte-offset of a __shared__ object (generic -> addrspace(3) -> int)
// ---------------------------------------------------------------------------
#define LDS_AS __attribute__((address_space(3)))
static __device__ __forceinline__ unsigned lds_addr_of(void* p) {
  return (unsigned)(unsigned long long)(LDS_AS void*)p;
}

// ---------------------------------------------------------------------------
// Issue one TDM 2D tile load: bf16 tile [tile_d1 rows x tile_d0 cols] from a
// row-major tensor with row stride `stride0` (elements) into LDS at lds_off.
// Rows beyond tensor_d1 / cols beyond tensor_d0 read as zero (TDM OOB rule).
// D# packing per CDNA5 ISA ch.8 (group0 128b, group1 256b; groups 2/3 unused
// for 2D tiles). 6-arg builtin form (clang-23 / therock headers toolchain).
// ---------------------------------------------------------------------------
static __device__ __forceinline__ void tdm_load_2d(
    unsigned lds_off, const __bf16* gptr,
    unsigned tensor_d0, unsigned tensor_d1,
    unsigned tile_d0, unsigned tile_d1,
    unsigned long stride0)
{
  unsigned long ga = (unsigned long)gptr;
  u32x4 g0;
  g0[0] = 1u;                                           // count=1 (valid D#)
  g0[1] = lds_off;                                      // lds_addr (bytes)
  g0[2] = (unsigned)(ga & 0xffffffffu);                 // global_addr[31:0]
  g0[3] = (unsigned)((ga >> 32) & 0x1ffffffu)           // global_addr[56:32]
          | (2u << 30);                                 // type = 2 ("image")
  i32x8 g1;
  g1[0] = (int)(1u << 16);                              // data_size=1 -> 2 bytes
  g1[1] = (int)((tensor_d0 & 0xffffu) << 16);           // tensor_dim0[15:0]
  g1[2] = (int)((tensor_d0 >> 16) | ((tensor_d1 & 0xffffu) << 16));
  g1[3] = (int)((tensor_d1 >> 16) | (tile_d0 << 16));   // tile_dim0
  g1[4] = (int)(tile_d1 & 0xffffu);                     // tile_dim1 (tile_dim2=0)
  g1[5] = (int)(unsigned)(stride0 & 0xffffffffu);       // tensor_dim0_stride lo
  g1[6] = (int)(unsigned)((stride0 >> 32) & 0xffffu);   // stride hi (dim1_stride=0)
  g1[7] = 0;
  i32x4 gz4; gz4[0] = 0; gz4[1] = 0; gz4[2] = 0; gz4[3] = 0;
  i32x8 gz8;
  gz8[0] = 0; gz8[1] = 0; gz8[2] = 0; gz8[3] = 0;
  gz8[4] = 0; gz8[5] = 0; gz8[6] = 0; gz8[7] = 0;
  __builtin_amdgcn_tensor_load_to_lds(g0, g1, gz4, gz4, gz8, 0);
}

// ---------------------------------------------------------------------------
// WMMA GEMM with TDM-staged, double-buffered LDS tiles.
//   C[z] = A1[z]@B1t[z]^T (+ A2[z]@B2t[z]^T) + bias[z] (+ resid[z])
// A:  bf16 [M,K] row-major.   Bt: bf16 [N,K] row-major (i.e. B pre-transposed).
// C:  f32 [M,N].  256 threads = 8 waves; block tile 128x128; K-step 32.
// Wave 0 drives the TDM pipeline (TENSORcnt); all waves compute.
// ---------------------------------------------------------------------------
__global__ __launch_bounds__(256)
void wmma_gemm_tdm(const __bf16* __restrict__ A1, const __bf16* __restrict__ B1t,
                   const __bf16* __restrict__ A2, const __bf16* __restrict__ B2t,
                   const float* __restrict__ bias, const float* __restrict__ resid,
                   float* __restrict__ C, int M, int N, int K,
                   long sA, long sB, long sBias, long sC)
{
  __shared__ __bf16 As[2][TM][TK];
  __shared__ __bf16 Bs[2][TN][TK];

  const long z = blockIdx.z;
  A1 += z * sA; B1t += z * sB;
  if (A2)    { A2   += z * sA; B2t += z * sB; }
  if (bias)  { bias  += z * sBias; }
  if (resid) { resid += z * sC; }
  C += z * sC;

  const int tid  = threadIdx.x;
  const int lane = tid & 31;
  const int wave = tid >> 5;
  const int half = lane >> 4;
  const int l16  = lane & 15;
  const int wm   = wave & 3;    // 4 waves along M (32 rows each)
  const int wn   = wave >> 2;   // 2 waves along N (64 cols each)
  const int row0 = blockIdx.x * TM;
  const int col0 = blockIdx.y * TN;

  const unsigned offA = lds_addr_of(&As[0][0][0]);
  const unsigned offB = lds_addr_of(&Bs[0][0][0]);
  const unsigned bufBytes = TM * TK * 2;   // 8 KB per buffer per matrix

  v8f acc[2][4];
  #pragma unroll
  for (int a = 0; a < 2; ++a)
    #pragma unroll
    for (int b = 0; b < 4; ++b)
      acc[a][b] = (v8f){0.f,0.f,0.f,0.f,0.f,0.f,0.f,0.f};

  for (int pass = 0; pass < 2; ++pass) {
    const __bf16* A  = pass ? A2  : A1;
    const __bf16* Bt = pass ? B2t : B1t;
    if (A == nullptr) continue;
    const int nk = K / TK;

    if (wave == 0) {   // prologue: stage tile 0 into buffer 0
      tdm_load_2d(offA, A  + (long)row0 * K, K, (unsigned)(M - row0), TK, TM, K);
      tdm_load_2d(offB, Bt + (long)col0 * K, K, (unsigned)(N - col0), TK, TN, K);
    }

    for (int kt = 0; kt < nk; ++kt) {
      const int cur = kt & 1;
      if (wave == 0) {
        if (kt + 1 < nk) {   // prefetch next tile into other buffer
          const int k1 = (kt + 1) * TK;
          tdm_load_2d(offA + (1 - cur) * bufBytes, A  + (long)row0 * K + k1,
                      K, (unsigned)(M - row0), TK, TM, K);
          tdm_load_2d(offB + (1 - cur) * bufBytes, Bt + (long)col0 * K + k1,
                      K, (unsigned)(N - col0), TK, TN, K);
          __builtin_amdgcn_s_wait_tensorcnt((short)2);  // current tile arrived
        } else {
          __builtin_amdgcn_s_wait_tensorcnt((short)0);
        }
      }
      __syncthreads();   // release LDS tile `cur` to all waves

      // ---- A fragments: contiguous k = 16*half + e per lane, same order as B ----
      v16bf afrag[2];
      #pragma unroll
      for (int sm = 0; sm < 2; ++sm) {
        const __bf16* ap = &As[cur][wm*32 + sm*16 + l16][0];
        v16bf f;
        #pragma unroll
        for (int e = 0; e < 8; ++e) f[e]   = ap[half*16 + e];
        #pragma unroll
        for (int e = 0; e < 8; ++e) f[8+e] = ap[half*16 + 8 + e];
        afrag[sm] = f;
      }
      // ---- B fragments + 8x WMMA ----
      #pragma unroll
      for (int sn = 0; sn < 4; ++sn) {
        const __bf16* bp = &Bs[cur][wn*64 + sn*16 + l16][half*16];
        v16bf bf;
        #pragma unroll
        for (int e = 0; e < 16; ++e) bf[e] = bp[e];
        #pragma unroll
        for (int sm = 0; sm < 2; ++sm)
          acc[sm][sn] = __builtin_amdgcn_wmma_f32_16x16x32_bf16(
              false, afrag[sm], false, bf, (short)0, acc[sm][sn], false, false);
      }
      __syncthreads();   // all waves done with tile `cur` before TDM reuses it
    }
  }

  // ---- epilogue: C/D f32 layout: VGPR v, lane -> M = v + 8*half, N = l16 ----
  #pragma unroll
  for (int sn = 0; sn < 4; ++sn) {
    int n = col0 + wn*64 + sn*16 + l16;
    if (n >= N) continue;
    float bv = bias ? bias[n] : 0.f;
    #pragma unroll
    for (int sm = 0; sm < 2; ++sm) {
      #pragma unroll
      for (int v = 0; v < 8; ++v) {
        int m = row0 + wm*32 + sm*16 + half*8 + v;
        if (m < M) {
          float val = acc[sm][sn][v] + bv;
          if (resid) val += resid[(long)m*N + n];
          C[(long)m*N + n] = val;
        }
      }
    }
  }
}

// ---------------------------------------------------------------------------
// helpers
// ---------------------------------------------------------------------------
__device__ __forceinline__ float block_sum(float v, float* red, int tid, int nthr) {
  red[tid] = v; __syncthreads();
  for (int s = nthr >> 1; s > 0; s >>= 1) {
    if (tid < s) red[tid] += red[tid + s];
    __syncthreads();
  }
  float r = red[0];
  __syncthreads();
  return r;
}

__global__ void zero_k(float* __restrict__ p, long n) {
  for (long i = blockIdx.x*(long)blockDim.x + threadIdx.x; i < n;
       i += (long)gridDim.x*blockDim.x) p[i] = 0.f;
}

__global__ void cvt_k(const float* __restrict__ s, __bf16* __restrict__ d, long n) {
  for (long i = blockIdx.x*(long)blockDim.x + threadIdx.x; i < n;
       i += (long)gridDim.x*blockDim.x) d[i] = (__bf16)s[i];
}

// Wt[n*K + k] = (bf16) W[k*N + n], batched over blockIdx.z
__global__ void cvt_t_k(const float* __restrict__ W, __bf16* __restrict__ Wt,
                        int K, int N, long sW, long sWt) {
  W  += (long)blockIdx.z * sW;
  Wt += (long)blockIdx.z * sWt;
  long total = (long)K * N;
  for (long i = blockIdx.x*(long)blockDim.x + threadIdx.x; i < total;
       i += (long)gridDim.x*blockDim.x) {
    long k = i / N, n = i % N;
    Wt[n*(long)K + k] = (__bf16)W[i];
  }
}

__global__ void vn_init_k(float* __restrict__ vne, const float* __restrict__ w, long n) {
  for (long i = blockIdx.x*(long)blockDim.x + threadIdx.x; i < n;
       i += (long)gridDim.x*blockDim.x) vne[i] = w[i % (NVN*HD)];
}

__global__ void count_k(const int* __restrict__ batch, float* __restrict__ counts, int n) {
  int i = blockIdx.x*blockDim.x + threadIdx.x;
  if (i < n) atomicAdd(&counts[batch[i]], 1.f);
}

__global__ void edge_scatter_k(const float* __restrict__ x, float* __restrict__ agg,
                               const int* __restrict__ src, const int* __restrict__ dst,
                               long E, int d) {
  long total = E * (long)d;
  for (long i = blockIdx.x*(long)blockDim.x + threadIdx.x; i < total;
       i += (long)gridDim.x*blockDim.x) {
    long e = i / d; int j = (int)(i % d);
    atomicAdd(&agg[(long)dst[e]*d + j], x[(long)src[e]*d + j]);
  }
}

__global__ void vn_mean_k(const float* __restrict__ vne, float* __restrict__ vm, long total) {
  for (long i = blockIdx.x*(long)blockDim.x + threadIdx.x; i < total;
       i += (long)gridDim.x*blockDim.x) {
    long g = i / HD; int c = (int)(i % HD);
    const float* b = vne + (g*NVN)*HD + c;
    vm[i] = (b[0] + b[HD] + b[2*HD]) * (1.f/3.f);
  }
}

// x_out = leaky_relu(LN(h[row] + vm[batch[row]])); writes f32 + bf16 copies
__global__ __launch_bounds__(256)
void node_post_k(const float* __restrict__ h, const float* __restrict__ vm,
                 const int* __restrict__ batch, const float* __restrict__ g,
                 const float* __restrict__ b, float* __restrict__ xout,
                 __bf16* __restrict__ xoutb) {
  __shared__ float red[256];
  const int row = blockIdx.x, tid = threadIdx.x;
  const float* hr = h  + (long)row * HD;
  const float* vr = vm + (long)batch[row] * HD;
  float vals[HD/256]; float s = 0.f;
  #pragma unroll
  for (int i = 0; i < HD/256; ++i) {
    int c = tid + i*256;
    vals[i] = hr[c] + vr[c]; s += vals[i];
  }
  float mu = block_sum(s, red, tid, 256) * (1.f/HD);
  float vs = 0.f;
  #pragma unroll
  for (int i = 0; i < HD/256; ++i) { float d = vals[i]-mu; vs += d*d; }
  float rstd = rsqrtf(block_sum(vs, red, tid, 256) * (1.f/HD) + LN_EPS);
  #pragma unroll
  for (int i = 0; i < HD/256; ++i) {
    int c = tid + i*256;
    float y = (vals[i]-mu)*rstd*g[c] + b[c];
    y = (y > 0.f) ? y : 0.01f*y;
    xout[(long)row*HD + c]  = y;
    xoutb[(long)row*HD + c] = (__bf16)y;
  }
}

__global__ void graph_accum_k(const float* __restrict__ x, const int* __restrict__ batch,
                              float* __restrict__ gm, long total) {
  for (long i = blockIdx.x*(long)blockDim.x + threadIdx.x; i < total;
       i += (long)gridDim.x*blockDim.x) {
    long n = i / HD; int c = (int)(i % HD);
    atomicAdd(&gm[(long)batch[n]*HD + c], x[i]);
  }
}

__global__ void gm_fin_k(float* __restrict__ gm, const float* __restrict__ counts, long total) {
  for (long i = blockIdx.x*(long)blockDim.x + threadIdx.x; i < total;
       i += (long)gridDim.x*blockDim.x) {
    long g = i / HD;
    gm[i] /= fmaxf(counts[g], 1.f);
  }
}

// vninb[v][g][c] = (bf16)(gm[g][c] + vne[g][v][c])
__global__ void vnin_k(const float* __restrict__ gm, const float* __restrict__ vne,
                       __bf16* __restrict__ vninb, long total) {
  for (long i = blockIdx.x*(long)blockDim.x + threadIdx.x; i < total;
       i += (long)gridDim.x*blockDim.x) {
    long v = i / ((long)GC*HD);
    long rem = i % ((long)GC*HD);
    long g = rem / HD; int c = (int)(rem % HD);
    vninb[i] = (__bf16)(gm[rem] + vne[(g*NVN + v)*HD + c]);
  }
}

// h1b = (bf16) relu(LN(h1f row, mg1[v], mB1[v])); rows = v*GC+g; width VH
__global__ __launch_bounds__(128)
void vn_h1_post_k(const float* __restrict__ h1f, __bf16* __restrict__ h1b,
                  const float* __restrict__ g1, const float* __restrict__ B1) {
  __shared__ float red[128];
  const int row = blockIdx.x, tid = threadIdx.x;
  const int v = row / GC;
  const float* p = h1f + (long)row * VH;
  float x = p[tid];
  float mu = block_sum(x, red, tid, 128) * (1.f/VH);
  float d = x - mu;
  float rstd = rsqrtf(block_sum(d*d, red, tid, 128) * (1.f/VH) + LN_EPS);
  float y = d * rstd * g1[v*VH + tid] + B1[v*VH + tid];
  h1b[(long)row*VH + tid] = (__bf16)fmaxf(y, 0.f);
}

// h2 post: relu(LN(row, mg2[v], mB2[v])); then vne[g][v] += result
__global__ __launch_bounds__(256)
void vn_h2_post_k(const float* __restrict__ h2, const float* __restrict__ g2,
                  const float* __restrict__ B2, float* __restrict__ vne) {
  __shared__ float red[256];
  const int row = blockIdx.x, tid = threadIdx.x;
  const int v = row / GC, g = row % GC;
  const float* p  = h2 + (long)row * HD;
  const float* gp = g2 + (long)v * HD;
  const float* bp = B2 + (long)v * HD;
  float vals[HD/256]; float s = 0.f;
  #pragma unroll
  for (int i = 0; i < HD/256; ++i) { vals[i] = p[tid + i*256]; s += vals[i]; }
  float mu = block_sum(s, red, tid, 256) * (1.f/HD);
  float vs = 0.f;
  #pragma unroll
  for (int i = 0; i < HD/256; ++i) { float d = vals[i]-mu; vs += d*d; }
  float rstd = rsqrtf(block_sum(vs, red, tid, 256) * (1.f/HD) + LN_EPS);
  #pragma unroll
  for (int i = 0; i < HD/256; ++i) {
    int c = tid + i*256;
    float y = (vals[i]-mu)*rstd*gp[c] + bp[c];
    vne[((long)g*NVN + v)*HD + c] += fmaxf(y, 0.f);
  }
}

// ---------------------------------------------------------------------------
extern "C" void kernel_launch(void* const* d_in, const int* in_sizes, int n_in,
                              void* d_out, int out_size, void* d_ws, size_t ws_size,
                              hipStream_t stream) {
  const float* x_in   = (const float*)d_in[0];
  const float* W0r    = (const float*)d_in[1];
  const float* W0a    = (const float*)d_in[2];
  const float* b0     = (const float*)d_in[3];
  const float* Wr_all = (const float*)d_in[4];
  const float* Wa_all = (const float*)d_in[5];
  const float* b_all  = (const float*)d_in[6];
  const float* ln_g   = (const float*)d_in[7];
  const float* ln_b   = (const float*)d_in[8];
  const float* vn_w   = (const float*)d_in[9];
  const float* mW1    = (const float*)d_in[10];
  const float* mb1    = (const float*)d_in[11];
  const float* mg1    = (const float*)d_in[12];
  const float* mB1    = (const float*)d_in[13];
  const float* mW2    = (const float*)d_in[14];
  const float* mb2    = (const float*)d_in[15];
  const float* mg2    = (const float*)d_in[16];
  const float* mB2    = (const float*)d_in[17];
  const float* W_out  = (const float*)d_in[18];
  const float* b_out  = (const float*)d_in[19];
  const int*   eidx   = (const int*)d_in[20];
  const int*   batch  = (const int*)d_in[21];

  const int  Nn = in_sizes[0] / INDIM;    // 10000
  const long E  = in_sizes[20] / 2;       // 80000

  // ---- workspace carve (256B aligned chunks) ----
  char* base = (char*)d_ws; size_t off = 0;
  auto alloc = [&](size_t bytes) -> void* {
    void* p = base + off; off += (bytes + 255) & ~(size_t)255; return p;
  };
  float* xbuf   = (float*)alloc((size_t)Nn*HD*4);
  float* hbuf   = (float*)alloc((size_t)Nn*HD*4);
  float* agg    = (float*)alloc((size_t)Nn*HD*4);
  float* vne    = (float*)alloc((size_t)GC*NVN*HD*4);
  float* vmean  = (float*)alloc((size_t)GC*HD*4);
  float* gm     = (float*)alloc((size_t)GC*HD*4);
  float* h1f    = (float*)alloc((size_t)NVN*GC*VH*4);
  float* h2     = (float*)alloc((size_t)NVN*GC*HD*4);
  float* counts = (float*)alloc(256*4);
  __bf16* xinb  = (__bf16*)alloc((size_t)Nn*INDIM*2);
  __bf16* xb    = (__bf16*)alloc((size_t)Nn*HD*2);
  __bf16* aggb  = (__bf16*)alloc((size_t)Nn*HD*2);
  __bf16* vninb = (__bf16*)alloc((size_t)NVN*GC*HD*2);
  __bf16* h1b   = (__bf16*)alloc((size_t)NVN*GC*VH*2);
  __bf16* w0r_t = (__bf16*)alloc((size_t)INDIM*HD*2);
  __bf16* w0a_t = (__bf16*)alloc((size_t)INDIM*HD*2);
  __bf16* wr_t  = (__bf16*)alloc((size_t)3*HD*HD*2);
  __bf16* wa_t  = (__bf16*)alloc((size_t)3*HD*HD*2);
  __bf16* mW1_t = (__bf16*)alloc((size_t)9*HD*VH*2);
  __bf16* mW2_t = (__bf16*)alloc((size_t)9*VH*HD*2);
  __bf16* wout_t= (__bf16*)alloc((size_t)HD*OUTDIM*2);
  (void)ws_size; (void)n_in; (void)out_size;

  const int EW = 2048;

  // ---- one-time prep: counts, vn_emb, bf16 weight transposes ----
  zero_k<<<1, 256, 0, stream>>>(counts, GC);
  count_k<<<(Nn+255)/256, 256, 0, stream>>>(batch, counts, Nn);
  vn_init_k<<<EW, 256, 0, stream>>>(vne, vn_w, (long)GC*NVN*HD);
  cvt_k<<<EW, 256, 0, stream>>>(x_in, xinb, (long)Nn*INDIM);
  cvt_t_k<<<dim3(256,1,1), 256, 0, stream>>>(W0r,   w0r_t, INDIM, HD, 0, 0);
  cvt_t_k<<<dim3(256,1,1), 256, 0, stream>>>(W0a,   w0a_t, INDIM, HD, 0, 0);
  cvt_t_k<<<dim3(1024,1,3),256, 0, stream>>>(Wr_all, wr_t, HD, HD, (long)HD*HD, (long)HD*HD);
  cvt_t_k<<<dim3(1024,1,3),256, 0, stream>>>(Wa_all, wa_t, HD, HD, (long)HD*HD, (long)HD*HD);
  cvt_t_k<<<dim3(256,1,9), 256, 0, stream>>>(mW1, mW1_t, HD, VH, (long)HD*VH, (long)HD*VH);
  cvt_t_k<<<dim3(256,1,9), 256, 0, stream>>>(mW2, mW2_t, VH, HD, (long)VH*HD, (long)VH*HD);
  cvt_t_k<<<dim3(64,1,1),  256, 0, stream>>>(W_out, wout_t, HD, OUTDIM, 0, 0);

  const float*  xcur  = x_in;
  const __bf16* xcurb = xinb;
  int dcur = INDIM;

  for (int i = 0; i < 4; ++i) {
    // agg = segment_sum(x[src] -> dst); then bf16 copy for the GEMM
    zero_k<<<EW, 256, 0, stream>>>(agg, (long)Nn*dcur);
    edge_scatter_k<<<EW, 256, 0, stream>>>(xcur, agg, eidx, eidx + E, E, dcur);
    cvt_k<<<EW, 256, 0, stream>>>(agg, aggb, (long)Nn*dcur);

    const __bf16* Wrt  = (i == 0) ? w0r_t : wr_t + (long)(i-1)*HD*HD;
    const __bf16* Wat  = (i == 0) ? w0a_t : wa_t + (long)(i-1)*HD*HD;
    const float*  bias = (i == 0) ? b0    : b_all + (long)(i-1)*HD;

    dim3 ggrid((Nn + TM - 1)/TM, HD/TN, 1);
    wmma_gemm_tdm<<<ggrid, 256, 0, stream>>>(
        xcurb, Wrt, aggb, Wat, bias, (i > 0) ? xcur : nullptr,
        hbuf, Nn, HD, dcur, 0, 0, 0, 0);

    vn_mean_k<<<EW, 256, 0, stream>>>(vne, vmean, (long)GC*HD);
    node_post_k<<<Nn, 256, 0, stream>>>(hbuf, vmean, batch,
                                        ln_g + (long)i*HD, ln_b + (long)i*HD,
                                        xbuf, xb);
    xcur = xbuf; xcurb = xb; dcur = HD;

    if (i < 3) {
      zero_k<<<EW, 256, 0, stream>>>(gm, (long)GC*HD);
      graph_accum_k<<<EW, 256, 0, stream>>>(xcur, batch, gm, (long)Nn*HD);
      gm_fin_k<<<EW, 256, 0, stream>>>(gm, counts, (long)GC*HD);
      vnin_k<<<EW, 256, 0, stream>>>(gm, vne, vninb, (long)NVN*GC*HD);

      // h1[v] = vnin[v] @ mW1[i][v] + mb1[i][v]   (M=64, N=128, K=1024)
      wmma_gemm_tdm<<<dim3(1,1,NVN), 256, 0, stream>>>(
          vninb, mW1_t + (long)i*NVN*HD*VH, nullptr, nullptr,
          mb1 + (long)i*NVN*VH, nullptr,
          h1f, GC, VH, HD,
          (long)GC*HD, (long)HD*VH, (long)VH, (long)GC*VH);
      vn_h1_post_k<<<NVN*GC, 128, 0, stream>>>(h1f, h1b,
                                               mg1 + (long)i*NVN*VH,
                                               mB1 + (long)i*NVN*VH);

      // h2[v] = h1[v] @ mW2[i][v] + mb2[i][v]     (M=64, N=1024, K=128)
      wmma_gemm_tdm<<<dim3(1, HD/TN, NVN), 256, 0, stream>>>(
          h1b, mW2_t + (long)i*NVN*VH*HD, nullptr, nullptr,
          mb2 + (long)i*NVN*HD, nullptr,
          h2, GC, HD, VH,
          (long)GC*VH, (long)VH*HD, (long)HD, (long)GC*HD);
      vn_h2_post_k<<<NVN*GC, 256, 0, stream>>>(h2, mg2 + (long)i*NVN*HD,
                                               mB2 + (long)i*NVN*HD, vne);
    }
  }

  // out = x @ W_out + b_out   (M=10000, N=40, K=1024)
  wmma_gemm_tdm<<<dim3((Nn + TM - 1)/TM, 1, 1), 256, 0, stream>>>(
      xb, wout_t, nullptr, nullptr, b_out, nullptr,
      (float*)d_out, Nn, OUTDIM, HD, 0, 0, 0, 0);
}
